// DifferentiableWorld_70884140253422
// MI455X (gfx1250) — matrix operand order repaired and checked by reading, not compiled, not run on hardware
//
#include <hip/hip_runtime.h>
#include <math.h>

// ---------------- problem constants ----------------
#define SRf 24000.0f
#define PI_D 3.14159265358979323846
constexpr int NFFT = 1024;
constexpr int HOP  = 256;
constexpr int HALF = 512;
constexpr int F    = 513;      // rfft bins
constexpr int Bb   = 8;
constexpr int T    = 4000;
constexpr int NS   = T * HOP;  // 1,024,000 samples
constexpr int R    = Bb * T;   // 32,000 frames
constexpr int KP   = 544;      // 513 padded to multiple of 32 (GEMM K)
constexpr int NP   = 544;      // 513 padded to multiple of 32 (GEMM N, 2-wide tile blocking)
constexpr int K2   = 1088;     // 2*NP  (Re|Im packed K)
constexpr int NW   = 512;      // synthesized samples per frame kept
constexpr int CH   = 4096;     // scan chunk
constexpr int NCH  = NS / CH;  // 250 chunks per row

typedef unsigned short u16;
typedef __attribute__((ext_vector_type(16))) __bf16 v16bf;
typedef __attribute__((ext_vector_type(8)))  float  v8f;

struct alignas(16) U4 { unsigned int a, b, c, d; };
union FragU { U4 q[2]; v16bf v; };
union AccU  { v8f v; float f[8]; };

// split f32 into bf16 hi + bf16 lo residual (≈fp32 via 3 bf16 WMMAs)
__device__ __forceinline__ void split_bf16(float x, u16* hi, u16* lo) {
  unsigned int bx = __float_as_uint(x);
  unsigned int h  = bx & 0xFFFF0000u;
  *hi = (u16)(h >> 16);
  float r = x - __uint_as_float(h);
  *lo = (u16)(__float_as_uint(r) >> 16);
}

// ---------------- split-bf16 GEMM:  C[M,N] = A[M,K] * B[K,N],  B given transposed [N,K]
// 2x2 register-blocked: each wave computes a 32x32 output block (4 WMMA tiles),
// reusing each A/B fragment twice -> 12 WMMAs per 16 b128 loads per 32-K step.
// A row-major bf16(hi/lo), K mult of 32; M mult of 32; N mult of 32; BT row-major [N][K].
__global__ __launch_bounds__(256) void gemm_bf16x3(
    const u16* __restrict__ Ahi, const u16* __restrict__ Alo,
    const u16* __restrict__ BThi, const u16* __restrict__ BTlo,
    float* __restrict__ C, int Mrows, int Kdim, int Ncols)
{
  int wave = (blockIdx.x * blockDim.x + threadIdx.x) >> 5;
  int lane = threadIdx.x & 31;
  int nt2 = Ncols >> 5;                  // N / 32
  int mt2 = Mrows >> 5;                  // M / 32
  int tm = wave / nt2, tn = wave % nt2;
  if (tm >= mt2) return;                 // wave-uniform: EXEC stays all-ones for WMMA
  int half = lane >> 4;                  // 0: lanes 0-15, 1: lanes 16-31
  int l16  = lane & 15;

  // A 16x32 bf16 layout: lane half h, VGPR0-3 -> K = k0+8h+0..7 ; VGPR4-7 -> K = k0+16+8h+0..7
  const u16* aH0 = Ahi + (size_t)(tm * 32 + l16) * Kdim + half * 8;
  const u16* aL0 = Alo + (size_t)(tm * 32 + l16) * Kdim + half * 8;
  const u16* aH1 = aH0 + (size_t)16 * Kdim;
  const u16* aL1 = aL0 + (size_t)16 * Kdim;
  // B 32x16 bf16 layout: lane = column, half h: VGPR0-3 -> K = k0+8h+0..7 ; VGPR4-7 -> K = k0+16+8h+0..7
  const u16* bH0 = BThi + (size_t)(tn * 32 + l16) * Kdim + half * 8;
  const u16* bL0 = BTlo + (size_t)(tn * 32 + l16) * Kdim + half * 8;
  const u16* bH1 = bH0 + (size_t)16 * Kdim;
  const u16* bL1 = bL0 + (size_t)16 * Kdim;

  v8f a00 = {}, a01 = {}, a10 = {}, a11 = {};
  for (int k0 = 0; k0 < Kdim; k0 += 32) {
    FragU fa0h, fa0l, fa1h, fa1l, fb0h, fb0l, fb1h, fb1l;
    fa0h.q[0] = *(const U4*)(aH0 + k0);  fa0h.q[1] = *(const U4*)(aH0 + k0 + 16);
    fa0l.q[0] = *(const U4*)(aL0 + k0);  fa0l.q[1] = *(const U4*)(aL0 + k0 + 16);
    fa1h.q[0] = *(const U4*)(aH1 + k0);  fa1h.q[1] = *(const U4*)(aH1 + k0 + 16);
    fa1l.q[0] = *(const U4*)(aL1 + k0);  fa1l.q[1] = *(const U4*)(aL1 + k0 + 16);
    fb0h.q[0] = *(const U4*)(bH0 + k0);  fb0h.q[1] = *(const U4*)(bH0 + k0 + 16);
    fb0l.q[0] = *(const U4*)(bL0 + k0);  fb0l.q[1] = *(const U4*)(bL0 + k0 + 16);
    fb1h.q[0] = *(const U4*)(bH1 + k0);  fb1h.q[1] = *(const U4*)(bH1 + k0 + 16);
    fb1l.q[0] = *(const U4*)(bL1 + k0);  fb1l.q[1] = *(const U4*)(bL1 + k0 + 16);

    a00 = __builtin_amdgcn_wmma_f32_16x16x32_bf16(false, fa0h.v, false, fb0h.v, (short)0, a00, false, false);
    a00 = __builtin_amdgcn_wmma_f32_16x16x32_bf16(false, fa0h.v, false, fb0l.v, (short)0, a00, false, false);
    a00 = __builtin_amdgcn_wmma_f32_16x16x32_bf16(false, fa0l.v, false, fb0h.v, (short)0, a00, false, false);

    a01 = __builtin_amdgcn_wmma_f32_16x16x32_bf16(false, fa0h.v, false, fb1h.v, (short)0, a01, false, false);
    a01 = __builtin_amdgcn_wmma_f32_16x16x32_bf16(false, fa0h.v, false, fb1l.v, (short)0, a01, false, false);
    a01 = __builtin_amdgcn_wmma_f32_16x16x32_bf16(false, fa0l.v, false, fb1h.v, (short)0, a01, false, false);

    a10 = __builtin_amdgcn_wmma_f32_16x16x32_bf16(false, fa1h.v, false, fb0h.v, (short)0, a10, false, false);
    a10 = __builtin_amdgcn_wmma_f32_16x16x32_bf16(false, fa1h.v, false, fb0l.v, (short)0, a10, false, false);
    a10 = __builtin_amdgcn_wmma_f32_16x16x32_bf16(false, fa1l.v, false, fb0h.v, (short)0, a10, false, false);

    a11 = __builtin_amdgcn_wmma_f32_16x16x32_bf16(false, fa1h.v, false, fb1h.v, (short)0, a11, false, false);
    a11 = __builtin_amdgcn_wmma_f32_16x16x32_bf16(false, fa1h.v, false, fb1l.v, (short)0, a11, false, false);
    a11 = __builtin_amdgcn_wmma_f32_16x16x32_bf16(false, fa1l.v, false, fb1h.v, (short)0, a11, false, false);
  }

  // C/D layout: lane col = l16, VGPR r -> row r + (half?8:0)
  int rb0 = tm * 32 + (half ? 8 : 0);
  int cb0 = tn * 32 + l16;
  AccU u0, u1, u2, u3;
  u0.v = a00; u1.v = a01; u2.v = a10; u3.v = a11;
  #pragma unroll
  for (int r2 = 0; r2 < 8; ++r2) {
    C[(size_t)(rb0 + r2) * Ncols + cb0]           = u0.f[r2];
    C[(size_t)(rb0 + r2) * Ncols + cb0 + 16]      = u1.f[r2];
    C[(size_t)(rb0 + 16 + r2) * Ncols + cb0]      = u2.f[r2];
    C[(size_t)(rb0 + 16 + r2) * Ncols + cb0 + 16] = u3.f[r2];
  }
}

// ---------------- setup kernels: fixed transform matrices (bf16 hi/lo, transposed layouts)

// MT[q][m], q<NP output phase bin, m<KP input log-mag bin.
// minphase[q] = sum_m logs[m]*M[m][q],  M = -(w_m*2/1024) * sum_{n=1}^{511} cos(2pi m n/1024) sin(2pi q n/1024)
__global__ void k_build_M(u16* MThi, u16* MTlo) {
  int q = blockIdx.y;
  int m = blockIdx.x * blockDim.x + threadIdx.x;
  if (m >= KP) return;
  float val = 0.f;
  if (m <= HALF && q <= HALF) {
    double am = (2.0 * PI_D * m) / 1024.0, aq = (2.0 * PI_D * q) / 1024.0;
    double cam = cos(am), sam = sin(am), caq = cos(aq), saq = sin(aq);
    double cm = cam, sm = sam, cq = caq, sq = saq;   // n = 1
    double acc = 0.0;
    for (int n = 1; n <= 511; ++n) {
      acc += cm * sq;
      double c1 = cm * cam - sm * sam, s1 = sm * cam + cm * sam; cm = c1; sm = s1;
      double c2 = cq * caq - sq * saq, s2 = sq * caq + cq * saq; cq = c2; sq = s2;
    }
    double wm = (m == 0 || m == HALF) ? 1.0 : 2.0;
    val = (float)(-(wm * 2.0 / 1024.0) * acc);
  }
  split_bf16(val, &MThi[(size_t)q * KP + m], &MTlo[(size_t)q * KP + m]);
}

// DT[q][j], q<K2, j<256 : rfft of 256-tap zero-padded frame. q<NP: cos; q>=NP: -sin.
__global__ void k_build_DT(u16* DThi, u16* DTlo) {
  int j = threadIdx.x;
  int q = blockIdx.y;
  double v = 0.0;
  if (q < NP) { if (q <= HALF) v = cos(2.0 * PI_D * q * j / 1024.0); }
  else { int q2 = q - NP; if (q2 <= HALF) v = -sin(2.0 * PI_D * q2 * j / 1024.0); }
  split_bf16((float)v, &DThi[(size_t)q * 256 + j], &DTlo[(size_t)q * 256 + j]);
}

// B2T[n][f], n<NW, f<K2 : irfft of hermitian spectrum, first 512 samples.
__global__ void k_build_B2T(u16* Bhi, u16* Blo) {
  int f = blockIdx.x * blockDim.x + threadIdx.x;
  int n = blockIdx.y;
  if (f >= K2) return;
  double v = 0.0;
  if (f < NP) {
    if (f <= HALF) { double wt = (f == 0 || f == HALF) ? 1.0 : 2.0;
                     v =  wt * cos(2.0 * PI_D * f * n / 1024.0) / 1024.0; }
  } else {
    int f2 = f - NP;
    if (f2 <= HALF) { double wt = (f2 == 0 || f2 == HALF) ? 1.0 : 2.0;
                      v = -wt * sin(2.0 * PI_D * f2 * n / 1024.0) / 1024.0; }
  }
  split_bf16((float)v, &Bhi[(size_t)n * K2 + f], &Blo[(size_t)n * K2 + f]);
}

// ---------------- F0 interp + hierarchical phase scan -> pulse train

__device__ __forceinline__ float f0_interp(const float* __restrict__ f0, int b, int n) {
  float pos = (n + 0.5f) * (1.0f / 256.0f) - 0.5f;     // T/N == 1/256
  pos = fminf(fmaxf(pos, 0.f), (float)(T - 1));
  int lo = (int)floorf(pos);
  int hi = min(lo + 1, T - 1);
  float w = pos - (float)lo;
  const float* fb = f0 + (size_t)b * T;
  return fb[lo] * (1.f - w) + fb[hi] * w;
}

__global__ __launch_bounds__(256) void k_f0_chunk_sums(const float* __restrict__ f0, double* csum) {
  int b = blockIdx.y, c = blockIdx.x, tid = threadIdx.x;
  int base = c * CH + tid * 16;
  double s = 0.0;
  for (int i = 0; i < 16; ++i) s += (double)f0_interp(f0, b, base + i) * (1.0 / 24000.0);
  __shared__ double sh[256];
  sh[tid] = s; __syncthreads();
  for (int off = 128; off; off >>= 1) { if (tid < off) sh[tid] += sh[tid + off]; __syncthreads(); }
  if (!tid) csum[(size_t)b * NCH + c] = sh[0];
}

// serial scan of chunk sums (tiny); optionally writes pref[row][NS] = total.
__global__ void k_offsets(const double* csum, double* coff, int nch, float* pref) {
  int row = blockIdx.x;
  if (threadIdx.x) return;
  const double* cs = csum + (size_t)row * nch;
  double* co = coff + (size_t)row * (nch + 1);
  double run = 0.0;
  for (int c = 0; c < nch; ++c) { co[c] = run; run += cs[c]; }
  co[nch] = run;
  if (pref) pref[(size_t)row * (NS + 1) + NS] = (float)run;
}

__global__ __launch_bounds__(256) void k_pulse(const float* __restrict__ f0,
                                               const double* __restrict__ coff,
                                               float* __restrict__ pulse) {
  int b = blockIdx.y, c = blockIdx.x, tid = threadIdx.x;
  int base = c * CH + tid * 16;
  float fv[16];
  double ts = 0.0;
  for (int i = 0; i < 16; ++i) { fv[i] = f0_interp(f0, b, base + i); ts += (double)fv[i] * (1.0 / 24000.0); }
  __shared__ double sh[256];
  sh[tid] = ts; __syncthreads();
  if (!tid) { double r = 0; for (int j = 0; j < 256; ++j) { double x = sh[j]; sh[j] = r; r += x; } }
  __syncthreads();
  double run = coff[(size_t)b * (NCH + 1) + c] + sh[tid];
  double pf = floor(run);
  float* pb = pulse + (size_t)b * NS;
  for (int i = 0; i < 16; ++i) {
    run += (double)fv[i] * (1.0 / 24000.0);
    double ph = floor(run);
    float t0p = SRf / fv[i];
    pb[base + i] = (float)(ph - pf) * sqrtf(fmaxf(t0p, 1.f));
    pf = ph;
  }
}

// ---------------- per-frame envelope/log staging (transpose [B,F,T] -> [R,*], pad K)

__global__ void k_env_logs(const float* __restrict__ env, const float* __restrict__ ap,
                           float* penv, float* aenv,
                           u16* lph, u16* lpl, u16* lah, u16* lal) {
  int r = blockIdx.y;
  int f = blockIdx.x * blockDim.x + threadIdx.x;
  if (f >= KP) return;
  int b = r / T, t = r - b * T;
  float lp = 0.f, la = 0.f;
  if (f < F) {
    size_t ix = ((size_t)b * F + f) * T + t;
    float e = env[ix], a = ap[ix];
    float pe = sqrtf(fmaxf(e * (1.f - a * a), 1e-20f));
    float ae = sqrtf(fmaxf(e * (a * a),       1e-20f));
    penv[(size_t)r * NP + f] = pe;
    aenv[(size_t)r * NP + f] = ae;
    lp = logf(fmaxf(pe, 1e-10f));
    la = logf(fmaxf(ae, 1e-10f));
  } else if (f < NP) {
    penv[(size_t)r * NP + f] = 0.f;
    aenv[(size_t)r * NP + f] = 0.f;
  }
  split_bf16(lp, &lph[(size_t)r * KP + f], &lpl[(size_t)r * KP + f]);
  split_bf16(la, &lah[(size_t)r * KP + f], &lal[(size_t)r * KP + f]);
}

__global__ void k_frames(const float* __restrict__ pulse, const float* __restrict__ noise,
                         u16* fph, u16* fpl, u16* fah, u16* fal) {
  int r = blockIdx.y;
  size_t ix = (size_t)r * 256 + threadIdx.x;   // (b*T+t)*256+j == b*NS + t*HOP + j
  split_bf16(pulse[ix], &fph[ix], &fpl[ix]);
  split_bf16(noise[ix], &fah[ix], &fal[ix]);
}

// P = env*exp(i*phi) * X   (elementwise complex), packed [Pr(0..NP-1) | Pi(0..NP-1)] in bf16 hi/lo
__global__ void k_make_pri(const float* __restrict__ penv, const float* __restrict__ phi,
                           const float* __restrict__ X, u16* ph, u16* pl) {
  int r = blockIdx.y;
  int f = blockIdx.x * blockDim.x + threadIdx.x;
  if (f >= NP) return;
  float Pr = 0.f, Pi = 0.f;
  if (f < F) {
    float pe = penv[(size_t)r * NP + f];
    float a  = phi [(size_t)r * NP + f];
    float s, c; sincosf(a, &s, &c);
    float cr = pe * c, si = pe * s;
    float Xr = X[(size_t)r * K2 + f], Xi = X[(size_t)r * K2 + NP + f];
    Pr = cr * Xr - si * Xi;
    Pi = cr * Xi + si * Xr;
  }
  split_bf16(Pr, &ph[(size_t)r * K2 + f],      &pl[(size_t)r * K2 + f]);
  split_bf16(Pi, &ph[(size_t)r * K2 + NP + f], &pl[(size_t)r * K2 + NP + f]);
}

// overlap-add of the shifted half-frames: sample s gets frame s/256 (n=s%256) and frame s/256-1 (n=s%256+256)
__global__ void k_ola(const float* __restrict__ W, float* __restrict__ wav) {
  int b = blockIdx.y;
  int s = blockIdx.x * blockDim.x + threadIdx.x;
  int t0 = s >> 8, n0 = s & 255;
  const float* Wb = W + (size_t)b * T * NW;
  float v = Wb[(size_t)t0 * NW + n0];
  if (t0 > 0) v += Wb[(size_t)(t0 - 1) * NW + n0 + 256];
  wav[(size_t)b * NS + s] = v;
}

// generic per-row chunk sums / prefix over [rows][NS] float data
__global__ __launch_bounds__(256) void k_chunk_sums(const float* __restrict__ data, double* csum) {
  int row = blockIdx.y, c = blockIdx.x, tid = threadIdx.x;
  const float* d = data + (size_t)row * NS + c * CH + tid * 16;
  double s = 0.0;
  for (int i = 0; i < 16; ++i) s += (double)d[i];
  __shared__ double sh[256];
  sh[tid] = s; __syncthreads();
  for (int off = 128; off; off >>= 1) { if (tid < off) sh[tid] += sh[tid + off]; __syncthreads(); }
  if (!tid) csum[(size_t)row * NCH + c] = sh[0];
}

__global__ __launch_bounds__(256) void k_prefix(const float* __restrict__ data,
                                                const double* __restrict__ coff,
                                                float* __restrict__ pref) {
  int row = blockIdx.y, c = blockIdx.x, tid = threadIdx.x;
  int base = c * CH + tid * 16;
  const float* d = data + (size_t)row * NS;
  float v[16]; double ts = 0.0;
  for (int i = 0; i < 16; ++i) { v[i] = d[base + i]; ts += (double)v[i]; }
  __shared__ double sh[256];
  sh[tid] = ts; __syncthreads();
  if (!tid) { double r = 0; for (int j = 0; j < 256; ++j) { double x = sh[j]; sh[j] = r; r += x; } }
  __syncthreads();
  double run = coff[(size_t)row * (NCH + 1) + c] + sh[tid];
  float* pr = pref + (size_t)row * (NS + 1);
  for (int i = 0; i < 16; ++i) { pr[base + i] = (float)run; run += (double)v[i]; }
}

// DC removal (centered 1024 moving average via prefix sums) + final output assembly
__global__ void k_dc_out(const float* __restrict__ wavPA, const float* __restrict__ prefPA,
                         float* __restrict__ out) {
  int b = blockIdx.y;
  int i = blockIdx.x * blockDim.x + threadIdx.x;
  int hi = min(i + 512, NS), lo = max(i - 512, 0);
  const float* prp = prefPA + (size_t)b * (NS + 1);
  const float* pra = prefPA + (size_t)(Bb + b) * (NS + 1);
  float p = wavPA[(size_t)b * NS + i]        - (prp[hi] - prp[lo]) * (1.f / 1024.f);
  float a = wavPA[(size_t)(Bb + b) * NS + i] - (pra[hi] - pra[lo]) * (1.f / 1024.f);
  out[(size_t)b * NS + i] = p + a;
  out[(size_t)Bb * NS + (size_t)b * NS + i] = a;
}

// ---------------- launcher ----------------
extern "C" void kernel_launch(void* const* d_in, const int* in_sizes, int n_in,
                              void* d_out, int out_size, void* d_ws, size_t ws_size,
                              hipStream_t stream) {
  const float* f0    = (const float*)d_in[0];   // [B,T]
  const float* env   = (const float*)d_in[1];   // [B,F,T]
  const float* ap    = (const float*)d_in[2];   // [B,F,T]
  const float* noise = (const float*)d_in[3];   // [B,N]
  float* out = (float*)d_out;

  char* p = (char*)d_ws;
  auto alloc = [&](size_t bytes) -> void* {
    void* r = (void*)p; p += (bytes + 255) & ~(size_t)255; return r;
  };

  float*  pulse  = (float*) alloc(sizeof(float)  * (size_t)R * 256);
  double* csum0  = (double*)alloc(sizeof(double) * (size_t)Bb * NCH);
  double* coff0  = (double*)alloc(sizeof(double) * (size_t)Bb * (NCH + 1));
  float*  penv   = (float*) alloc(sizeof(float)  * (size_t)R * NP);
  float*  aenv   = (float*) alloc(sizeof(float)  * (size_t)R * NP);
  char*   logsBase = p;                               // reused later for wav/pref
  u16* lph = (u16*)alloc(sizeof(u16) * (size_t)R * KP);
  u16* lpl = (u16*)alloc(sizeof(u16) * (size_t)R * KP);
  u16* lah = (u16*)alloc(sizeof(u16) * (size_t)R * KP);
  u16* lal = (u16*)alloc(sizeof(u16) * (size_t)R * KP);
  u16* MThi = (u16*)alloc(sizeof(u16) * (size_t)NP * KP);
  u16* MTlo = (u16*)alloc(sizeof(u16) * (size_t)NP * KP);
  char* phiBase = p;                                  // reused later for W
  float* PHI_P = (float*)alloc(sizeof(float) * (size_t)R * NP);
  float* PHI_A = (float*)alloc(sizeof(float) * (size_t)R * NP);
  u16* fph = (u16*)alloc(sizeof(u16) * (size_t)R * 256);
  u16* fpl = (u16*)alloc(sizeof(u16) * (size_t)R * 256);
  u16* fah = (u16*)alloc(sizeof(u16) * (size_t)R * 256);
  u16* fal = (u16*)alloc(sizeof(u16) * (size_t)R * 256);
  u16* DThi = (u16*)alloc(sizeof(u16) * (size_t)K2 * 256);
  u16* DTlo = (u16*)alloc(sizeof(u16) * (size_t)K2 * 256);
  float* XP = (float*)alloc(sizeof(float) * (size_t)R * K2);
  float* XA = (float*)alloc(sizeof(float) * (size_t)R * K2);
  u16* prPh = (u16*)alloc(sizeof(u16) * (size_t)R * K2);
  u16* prPl = (u16*)alloc(sizeof(u16) * (size_t)R * K2);
  u16* prAh = (u16*)alloc(sizeof(u16) * (size_t)R * K2);
  u16* prAl = (u16*)alloc(sizeof(u16) * (size_t)R * K2);
  u16* B2Thi = (u16*)alloc(sizeof(u16) * (size_t)NW * K2);
  u16* B2Tlo = (u16*)alloc(sizeof(u16) * (size_t)NW * K2);
  double* csumDC = (double*)alloc(sizeof(double) * 16 * NCH);
  double* coffDC = (double*)alloc(sizeof(double) * 16 * (NCH + 1));

  // region reuse (lifetimes disjoint): W (post min-phase) overlays PHI; wav/pref overlay logs.
  float* W     = (float*)phiBase;                     // [2][R][NW] : WP then WA
  float* WP = W;           float* WA = W + (size_t)R * NW;
  float* wavPA = (float*)logsBase;                    // [16][NS]
  float* prefPA = wavPA + (size_t)2 * Bb * NS;        // [16][NS+1]

  // --- setup transform matrices
  k_build_M  <<<dim3((KP + 255) / 256, NP), 256, 0, stream>>>(MThi, MTlo);
  k_build_DT <<<dim3(1, K2),              256, 0, stream>>>(DThi, DTlo);
  k_build_B2T<<<dim3((K2 + 255) / 256, NW), 256, 0, stream>>>(B2Thi, B2Tlo);

  // --- pulse train via hierarchical phase scan
  k_f0_chunk_sums<<<dim3(NCH, Bb), 256, 0, stream>>>(f0, csum0);
  k_offsets      <<<Bb, 32, 0, stream>>>(csum0, coff0, NCH, nullptr);
  k_pulse        <<<dim3(NCH, Bb), 256, 0, stream>>>(f0, coff0, pulse);

  // --- envelope staging + log-magnitudes
  k_env_logs<<<dim3((KP + 255) / 256, R), 256, 0, stream>>>(env, ap, penv, aenv, lph, lpl, lah, lal);
  k_frames  <<<dim3(1, R), 256, 0, stream>>>(pulse, noise, fph, fpl, fah, fal);

  auto launch_gemm = [&](const u16* Ah, const u16* Al, const u16* Bh, const u16* Bl,
                         float* C, int M, int K, int N) {
    int tiles  = (M / 32) * (N / 32);        // one wave per 32x32 block
    int blocks = (tiles + 7) / 8;            // 8 waves per 256-thread block
    gemm_bf16x3<<<blocks, 256, 0, stream>>>(Ah, Al, Bh, Bl, C, M, K, N);
  };

  // --- min-phase:  PHI = logs @ M          [32000 x 544] x [544 x 544]
  launch_gemm(lph, lpl, MThi, MTlo, PHI_P, R, KP, NP);
  launch_gemm(lah, lal, MThi, MTlo, PHI_A, R, KP, NP);

  // --- rfft of frames:  X = frames @ DFT   [32000 x 256] x [256 x 1088]
  launch_gemm(fph, fpl, DThi, DTlo, XP, R, 256, K2);
  launch_gemm(fah, fal, DThi, DTlo, XA, R, 256, K2);

  // --- complex filter (elementwise)
  k_make_pri<<<dim3((NP + 255) / 256, R), 256, 0, stream>>>(penv, PHI_P, XP, prPh, prPl);
  k_make_pri<<<dim3((NP + 255) / 256, R), 256, 0, stream>>>(aenv, PHI_A, XA, prAh, prAl);

  // --- irfft (first 512 samples):  W = [Pr|Pi] @ B2   [32000 x 1088] x [1088 x 512]
  launch_gemm(prPh, prPl, B2Thi, B2Tlo, WP, R, K2, NW);
  launch_gemm(prAh, prAl, B2Thi, B2Tlo, WA, R, K2, NW);

  // --- overlap-add
  k_ola<<<dim3(NS / 256, Bb), 256, 0, stream>>>(WP, wavPA);
  k_ola<<<dim3(NS / 256, Bb), 256, 0, stream>>>(WA, wavPA + (size_t)Bb * NS);

  // --- DC removal prefix sums + output
  k_chunk_sums<<<dim3(NCH, 16), 256, 0, stream>>>(wavPA, csumDC);
  k_offsets   <<<16, 32, 0, stream>>>(csumDC, coffDC, NCH, prefPA);
  k_prefix    <<<dim3(NCH, 16), 256, 0, stream>>>(wavPA, coffDC, prefPA);
  k_dc_out    <<<dim3(NS / 256, Bb), 256, 0, stream>>>(wavPA, prefPA, out);

  (void)in_sizes; (void)n_in; (void)out_size; (void)ws_size;
}
